// additive_intervention_30073361007228
// MI455X (gfx1250) — compile-verified, matrix operand order checked by reading, not run on hardware
//
#include <hip/hip_runtime.h>

// ---------------- problem sizes (fixed by the reference) ----------------
constexpr int NB    = 32;     // batch
constexpr int NN    = 8192;   // set size
constexpr int NCON  = 256;    // confounder dim (= output dim)
constexpr int NFUSE = 512;
constexpr int NHID  = 256;

// ---------------- tiling ----------------
constexpr int WAVES          = 8;                      // 8 waves / workgroup
constexpr int NTHREADS       = WAVES * 32;             // 256 (wave32!)
constexpr int STRIPS         = 16;                     // strips per batch row
constexpr int ROWS_PER_STRIP = NN / STRIPS;            // 512
constexpr int GROUP_ROWS     = WAVES * 16;             // 128 rows per group
constexpr int NGROUPS        = ROWS_PER_STRIP / GROUP_ROWS; // 4
constexpr int KCHUNKS        = NCON / 32;              // 8  (K=32 per wmma)
constexpr int HCHUNKS        = NHID / 16;              // 16 (N=16 per wmma)
constexpr int NFRAG          = HCHUNKS * KCHUNKS;      // 128 B-fragments
constexpr int WK_U4          = NFRAG * 64;             // 8192 uint4 = 128 KB

typedef __attribute__((ext_vector_type(16))) __bf16 v16bf;
typedef __attribute__((ext_vector_type(8)))  float  v8f;

__device__ __forceinline__ float fast_tanh(float x) {
  x = fminf(fmaxf(x, -15.f), 15.f);
  float e = __expf(2.f * x);
  return (e - 1.f) / (e + 1.f);
}

// ---------------------------------------------------------------
// Kernel 1: query[b,h] = sum_f fuse[b,f] * Wq[h,f]      [32,256]
// ---------------------------------------------------------------
__global__ void query_kernel(const float* __restrict__ fuse,
                             const float* __restrict__ Wq,
                             float* __restrict__ q) {
  int b = blockIdx.x;
  int h = threadIdx.x;
  const float* fr = fuse + b * NFUSE;
  const float* wr = Wq + (size_t)h * NFUSE;
  float acc = 0.f;
#pragma unroll 4
  for (int f = 0; f < NFUSE; ++f) acc += fr[f] * wr[f];
  q[b * NHID + h] = acc;
}

// ---------------------------------------------------------------
// Kernel 2: pack Wk (fp32, [HID=256, CON=256]) into bf16 B-matrix
// fragments for v_wmma_f32_16x16x32_bf16, in an LDS-bank-friendly
// order.
// Fragment (hk, kk): B[k][n] = Wk[hk*16 + n][kk*32 + k], k in 0..31.
// 16-bit B layout (mirror of documented A layout):
//   lane L: n = L%16, half = L/16
//   elem e (0..15): k = half*8 + (e < 8 ? e : e + 8)
// Bank-friendly linear bf16 index:
//   frag = hk*8 + kk
//   idx  = frag*512 + d*256 + L*8 + e_lo      (d = e/8, e_lo = e%8)
// so each ds_load_b128 (one 16B half per lane) covers 512 contiguous
// bytes across the 32 lanes -> all 64 LDS banks, no conflicts.
// ---------------------------------------------------------------
__global__ void pack_wk_kernel(const float* __restrict__ Wk,
                               unsigned short* __restrict__ pack_u) {
  __bf16* pack = reinterpret_cast<__bf16*>(pack_u);
  int idx  = blockIdx.x * blockDim.x + threadIdx.x;   // 0 .. 65535
  int e_lo = idx & 7;
  int L    = (idx >> 3) & 31;
  int d    = (idx >> 8) & 1;
  int frag = idx >> 9;
  int kk   = frag & 7;
  int hk   = frag >> 3;
  int h    = hk * 16 + (L & 15);
  int k    = (L >> 4) * 8 + e_lo + d * 16;
  int c    = kk * 32 + k;
  pack[idx] = (__bf16)Wk[(size_t)h * NCON + c];
}

// ---------------------------------------------------------------
// Kernel 3: fused keyp-GEMM + tanh + Wt-dot + online softmax +
// weighted conf accumulation. One pass over confounder_set.
// grid = NB*STRIPS blocks, 256 threads (8 waves).
// Wk (128 KB bf16, packed) is staged once per workgroup in LDS
// (CDNA5: 320 KB LDS per WGP) and feeds the WMMAs via ds_load_b128,
// software-pipelined one fragment ahead so LDS latency overlaps the
// WMMA accumulate chain.
// Per strip outputs: [m, s, fin[256]] (unnormalized, per-strip max).
// ---------------------------------------------------------------
__global__ void attn_kernel(const float* __restrict__ conf,
                            const float* __restrict__ q,
                            const float* __restrict__ Wt,
                            const unsigned short* __restrict__ wkpack_u,
                            float* __restrict__ strip_out) {
  __shared__ uint4 s_wk[WK_U4];       // 128 KB packed Wk (bf16)
  __shared__ float s_fin[NCON];
  __shared__ float s_logit[GROUP_ROWS];
  __shared__ float s_m, s_s;

  int b     = blockIdx.x / STRIPS;
  int strip = blockIdx.x % STRIPS;
  int tid   = threadIdx.x;
  int wave  = tid >> 5;
  int lane  = tid & 31;          // wave32 on gfx1250
  int mrow  = lane & 15;         // M index within 16-row tile
  int half  = lane >> 4;         // K-half select for 16-bit A/B layout

  // ---- one-time LDS fill: packed Wk (straight 128 KB copy) ----
  {
    const uint4* src = reinterpret_cast<const uint4*>(wkpack_u);
    for (int i = tid; i < WK_U4; i += NTHREADS) s_wk[i] = src[i];
  }
  for (int c = tid; c < NCON; c += NTHREADS) s_fin[c] = 0.f;
  if (tid == 0) { s_m = -1e30f; s_s = 0.f; }
  __syncthreads();

  const float* qb = q + b * NHID;

  for (int g = 0; g < NGROUPS; ++g) {
    int nbase = strip * ROWS_PER_STRIP + g * GROUP_ROWS;
    int row   = nbase + wave * 16 + mrow;
    const float* rowp = conf + ((size_t)b * NN + row) * NCON;

    // prefetch next group's row for this lane (global_prefetch_b8)
    if (g + 1 < NGROUPS)
      __builtin_prefetch(rowp + (size_t)GROUP_ROWS * NCON, 0, 1);

    // ---- load A tile (16 x 256 fp32) as 8 bf16 fragments ----
    // lane holds row `mrow`; K groups: half*8 + [0..7] and half*8 + [16..23]
    v16bf a[KCHUNKS];
#pragma unroll
    for (int kk = 0; kk < KCHUNKS; ++kk) {
      int c0 = kk * 32 + half * 8;
      float tmp[16];
      ((float4*)tmp)[0] = *(const float4*)(rowp + c0);
      ((float4*)tmp)[1] = *(const float4*)(rowp + c0 + 4);
      ((float4*)tmp)[2] = *(const float4*)(rowp + c0 + 16);
      ((float4*)tmp)[3] = *(const float4*)(rowp + c0 + 20);
#pragma unroll
      for (int e = 0; e < 16; ++e) a[kk][e] = (__bf16)tmp[e];
    }

    // ---- per-lane logit partials over all h ----
    float lacc[8];
#pragma unroll
    for (int r = 0; r < 8; ++r) lacc[r] = 0.f;

    // B-fragment double buffer over the flattened fragment stream
    // (frag = hk*KCHUNKS + kk); prefetch frag+1 across hk boundaries
    // so only the first fragment per group exposes LDS latency.
    union BB { uint4 u[2]; v16bf v; };
    BB cur, nxt;
    cur.u[0] = s_wk[lane];
    cur.u[1] = s_wk[lane + 32];

    for (int hk = 0; hk < HCHUNKS; ++hk) {
      // C/D layout: VGPR r, lane L -> M = r + 8*half, Nd = L%16.
      // So each lane sees a single h = hk*16 + mrow across all 8 acc regs.
      float wt = Wt[hk * 16 + mrow];
      float qv = qb[hk * 16 + mrow];

      v8f acc = {};
#pragma unroll
      for (int kk = 0; kk < KCHUNKS; ++kk) {
        int fn = hk * KCHUNKS + kk + 1;     // next fragment (may cross hk)
        if (fn < NFRAG) {
          nxt.u[0] = s_wk[fn * 64 + lane];       // ds_load_b128 (issued early)
          nxt.u[1] = s_wk[fn * 64 + 32 + lane];  // ds_load_b128 (issued early)
        }
        acc = __builtin_amdgcn_wmma_f32_16x16x32_bf16(
            false, a[kk], false, cur.v, (short)0, acc, false, false);
        if (fn < NFRAG) cur = nxt;
      }
#pragma unroll
      for (int r = 0; r < 8; ++r)
        lacc[r] += fast_tanh(qv + acc[r]) * wt;
    }

    // reduce over the 16 h-lanes within each half (h-sum)
#pragma unroll
    for (int r = 0; r < 8; ++r) {
      float v = lacc[r];
      v += __shfl_xor(v, 1);
      v += __shfl_xor(v, 2);
      v += __shfl_xor(v, 4);
      v += __shfl_xor(v, 8);
      lacc[r] = v;
    }
    if (mrow == 0) {
      // half 0 -> rows 0..7, half 1 -> rows 8..15 of this wave's tile
#pragma unroll
      for (int r = 0; r < 8; ++r)
        s_logit[wave * 16 + half * 8 + r] = lacc[r];
    }
    __syncthreads();

    // ---- online softmax + weighted accumulation of conf rows ----
    float m_old = s_m;
    float gmax = -1e30f;
    for (int i = 0; i < GROUP_ROWS; ++i) gmax = fmaxf(gmax, s_logit[i]);
    float newm  = fmaxf(m_old, gmax);
    float scale = __expf(m_old - newm);

    int c0 = tid;                           // 256 threads x 1 col = 256
    float f0 = s_fin[c0] * scale;
    float esum = 0.f;
    const float* cp = conf + ((size_t)b * NN + nbase) * NCON + c0;
    for (int i = 0; i < GROUP_ROWS; ++i) {
      float e = __expf(s_logit[i] - newm);
      esum += e;
      f0 += e * cp[(size_t)i * NCON];       // WGP$-hot, fully coalesced
    }
    s_fin[c0] = f0;
    __syncthreads();
    if (tid == 0) { s_s = s_s * scale + esum; s_m = newm; }
    __syncthreads();
  }

  // ---- write per-strip partials: [m, s, fin[256]] ----
  float* out = strip_out + (size_t)(b * STRIPS + strip) * (NCON + 2);
  if (tid == 0) { out[0] = s_m; out[1] = s_s; }
  for (int c = tid; c < NCON; c += NTHREADS) out[2 + c] = s_fin[c];
}

// ---------------------------------------------------------------
// Kernel 4: combine strips -> fin[b,c] (fp32 output)
// ---------------------------------------------------------------
__global__ void combine_kernel(const float* __restrict__ strip_out,
                               float* __restrict__ out) {
  int b = blockIdx.x;
  int c = threadIdx.x;
  float M = -1e30f;
#pragma unroll
  for (int k = 0; k < STRIPS; ++k)
    M = fmaxf(M, strip_out[(size_t)(b * STRIPS + k) * (NCON + 2)]);
  float S = 0.f, F = 0.f;
#pragma unroll
  for (int k = 0; k < STRIPS; ++k) {
    const float* p = strip_out + (size_t)(b * STRIPS + k) * (NCON + 2);
    float w = __expf(p[0] - M);
    S += w * p[1];
    F += w * p[2 + c];
  }
  out[b * NCON + c] = F / S;
}

// ---------------------------------------------------------------
extern "C" void kernel_launch(void* const* d_in, const int* in_sizes, int n_in,
                              void* d_out, int out_size, void* d_ws, size_t ws_size,
                              hipStream_t stream) {
  const float* conf = (const float*)d_in[0];  // [32, 8192, 256] f32
  const float* fuse = (const float*)d_in[1];  // [32, 512] f32
  const float* Wq   = (const float*)d_in[2];  // [256, 512] f32
  const float* Wk   = (const float*)d_in[3];  // [256, 256] f32
  const float* Wt   = (const float*)d_in[4];  // [1, 256] f32
  float* out = (float*)d_out;                 // [32, 256] f32

  char* ws = (char*)d_ws;
  float*          q      = (float*)ws;                          // 8192 f32 = 32 KB
  unsigned short* wkpack = (unsigned short*)(ws + 32768);       // 65536 bf16 = 128 KB
  float*          strip  = (float*)(ws + 32768 + 131072);       // 32*16*258 f32 ≈ 516 KB

  hipLaunchKernelGGL(query_kernel,   dim3(NB),          dim3(NHID),     0, stream,
                     fuse, Wq, q);
  hipLaunchKernelGGL(pack_wk_kernel, dim3(256),         dim3(256),      0, stream,
                     Wk, wkpack);
  hipLaunchKernelGGL(attn_kernel,    dim3(NB * STRIPS), dim3(NTHREADS), 0, stream,
                     conf, q, Wt, wkpack, strip);
  hipLaunchKernelGGL(combine_kernel, dim3(NB),          dim3(NCON),     0, stream,
                     strip, out);
}
